// Qwen2_5_VL_VisionModel_4209067950452
// MI455X (gfx1250) — matrix operand order repaired and checked by reading, not compiled, not run on hardware
//
#include <hip/hip_runtime.h>
#include <math.h>

// ---------------------------------------------------------------------------
// Qwen2.5-VL vision tower forward for MI455X (gfx1250, wave32, WMMA).
// GEMMs: f32 in HBM, f32->bf16 convert during staging (global_load_b128 ->
// regs -> ds_store_b128/b32), v_wmma_f32_16x16x32_bf16 w/ f32 accum, fused
// epilogues, register-pipelined global loads overlapping WMMA.
// Templated on TRANSB and epilogue MODE: branch-free hot loops.
// ---------------------------------------------------------------------------

typedef __attribute__((ext_vector_type(16))) __bf16 v16bf;
typedef __attribute__((ext_vector_type(8)))  __bf16 v8bf;
typedef __attribute__((ext_vector_type(2)))  __bf16 v2bf;
typedef __attribute__((ext_vector_type(8)))  float  v8f;

#define T_TOK    1024
#define DMODEL   1280
#define NHEAD    16
#define HDIM     80
#define INTER_F  3456
#define WINDOW_SZ 64

__device__ __forceinline__ float sigmoidf_(float x) { return 1.f / (1.f + __expf(-x)); }

// Load 16 consecutive floats at k0..k0+15 from a (clamped) row base.
// Fast path: 4x float4 (global_load_b128). K-tail: clamped loads + select 0.
__device__ __forceinline__ void load_row16(float (&r)[16], const float* base,
                                           int k0, int K)
{
    if (k0 + 16 <= K) {
        const float4* p4 = (const float4*)(base + k0);
#pragma unroll
        for (int i = 0; i < 4; i++) {
            float4 t = p4[i];
            r[4*i+0] = t.x; r[4*i+1] = t.y; r[4*i+2] = t.z; r[4*i+3] = t.w;
        }
    } else {
#pragma unroll
        for (int i = 0; i < 16; i++) {
            int kk = k0 + i;
            int kc = (kk < K) ? kk : (K - 1);     // clamped, always-valid load
            float v = base[kc];
            r[i] = (kk < K) ? v : 0.f;            // branchless zero-fill
        }
    }
}

// Non-transposed B: two k-rows (kk0, kk0+1) x 8 consecutive n.
// r[0..7] = row kk0, r[8..15] = row kk0+1. K-rows clamped/zeroed; cols clamped
// (clamped cols only feed masked output columns).
__device__ __forceinline__ void load_bnt_pair(float (&r)[16], const float* Bb,
                                              long ldb, int kk0, int n0,
                                              int N, int K)
{
#pragma unroll
    for (int hrow = 0; hrow < 2; hrow++) {
        int kk = kk0 + hrow;
        int kc = (kk < K) ? kk : (K - 1);
        const float* p = Bb + (long)kc * ldb;
        float* rr = &r[8 * hrow];
        if (n0 + 8 <= N) {
            const float4* p4 = (const float4*)(p + n0);
#pragma unroll
            for (int i = 0; i < 2; i++) {
                float4 t = p4[i];
                rr[4*i+0] = t.x; rr[4*i+1] = t.y; rr[4*i+2] = t.z; rr[4*i+3] = t.w;
            }
        } else {
#pragma unroll
            for (int i = 0; i < 8; i++) {
                int nn = n0 + i;
                rr[i] = p[(nn < N) ? nn : (N - 1)];
            }
        }
        if (kk >= K) {
#pragma unroll
            for (int i = 0; i < 8; i++) rr[i] = 0.f;
        }
    }
}

// ---------------------------------------------------------------------------
// Generic batched GEMM: C[z] = epilogue(A[z] @ B[z] + bias)
//   A: M x K row-major (lda), per-batch stride sA
//   B: TRANSB==0 -> K x N row-major; TRANSB==1 -> N x K row-major
//   MODE 0: acc (+bias)     MODE 1: res + acc + bias
//   MODE 2: gelu(acc+bias)  MODE 3: silu(acc+bias) * res
// 256 threads = 8 wave32 waves (2x4), block tile 128x128, wave tile 64x32,
// K-step 32, 8 x v_wmma_f32_16x16x32_bf16 per wave per K-step.
// ---------------------------------------------------------------------------
template <int TRANSB, int MODE>
__global__ __launch_bounds__(256)
void gemm_bf16_wmma(const float* __restrict__ A, long lda, long sA,
                    const float* __restrict__ B, long ldb, long sB,
                    float* __restrict__ C, long ldc, long sC,
                    const float* __restrict__ bias,
                    const float* __restrict__ res,
                    int M, int N, int K)
{
    __shared__ __align__(16) __bf16 As[128][40];   // [m][k], row stride 80B
    __shared__ __align__(16) __bf16 Bs[128][40];   // [n][k], K-contiguous

    const int tid  = threadIdx.x;
    const int lane = tid & 31;
    const int wave = tid >> 5;
    const int wm   = wave >> 2;          // 0..1
    const int wn   = wave & 3;           // 0..3
    const int bm   = blockIdx.y * 128;
    const int bn   = blockIdx.x * 128;
    const int z    = blockIdx.z;

    const float* Ab = A + (long)z * sA;
    const float* Bb = B + (long)z * sB;
    float*       Cb = C + (long)z * sC;

    // staging coordinates (row/col-clamped: garbage only feeds accumulator
    // entries that are masked at the final store; only K needs zero-fill)
    int arow = bm + (tid >> 1); if (arow >= M) arow = M - 1;
    const int akh = (tid & 1) * 16;
    const float* arow_p = Ab + (long)arow * lda;

    const float* brow_p = Bb;
    int bkp = 0, bn0 = 0;
    if (TRANSB) {
        int nrow = bn + (tid >> 1); if (nrow >= N) nrow = N - 1;
        brow_p = Bb + (long)nrow * ldb;
    } else {
        bkp = (tid >> 4) * 2;             // k-pair base: 0,2,..,30
        bn0 = bn + (tid & 15) * 8;        // 8 consecutive n
    }

    v8f acc[4][2];
#pragma unroll
    for (int i = 0; i < 4; i++)
#pragma unroll
        for (int j = 0; j < 2; j++)
#pragma unroll
            for (int e = 0; e < 8; e++) acc[i][j][e] = 0.f;

    float rA[16];
    float rB[16];

    // ---- prologue: K-tile 0 into registers ----
    load_row16(rA, arow_p, akh, K);
    if (TRANSB) load_row16(rB, brow_p, akh, K);
    else        load_bnt_pair(rB, Bb, ldb, bkp, bn0, N, K);

    for (int kt = 0; kt < K; kt += 32) {
        // ---- store current register tile to LDS (f32 -> bf16 packed) ----
        {
            v8bf lo, hi;
#pragma unroll
            for (int i = 0; i < 8; i++) { lo[i] = (__bf16)rA[i]; hi[i] = (__bf16)rA[8 + i]; }
            v8bf* d = (v8bf*)&As[tid >> 1][akh];
            d[0] = lo; d[1] = hi;
        }
        if (TRANSB) {
            v8bf lo, hi;
#pragma unroll
            for (int i = 0; i < 8; i++) { lo[i] = (__bf16)rB[i]; hi[i] = (__bf16)rB[8 + i]; }
            v8bf* d = (v8bf*)&Bs[tid >> 1][akh];
            d[0] = lo; d[1] = hi;
        } else {
            // k-pair packed transpose: 8x ds_store_b32, each cvt_pk fully used
            const int nloc = (tid & 15) * 8;
#pragma unroll
            for (int i = 0; i < 8; i++) {
                v2bf pk;
                pk[0] = (__bf16)rB[i];
                pk[1] = (__bf16)rB[8 + i];
                *(v2bf*)&Bs[nloc + i][bkp] = pk;
            }
        }
        __syncthreads();

        // ---- issue next tile's global loads (overlap with WMMA below) ----
        const int ktn = kt + 32;
        if (ktn < K) {
            load_row16(rA, arow_p, ktn + akh, K);
            if (TRANSB) load_row16(rB, brow_p, ktn + akh, K);
            else        load_bnt_pair(rB, Bb, ldb, ktn + bkp, bn0, N, K);
            if (ktn + 32 < K)   // global_prefetch_b8, two tiles ahead
                __builtin_prefetch(arow_p + ktn + 32 + akh, 0, 1);
        }

        // ---- fragments + WMMA ----
        // A frag: lane row = lane&15, kb=(lane>>4)*8, K in [kb,kb+8)+[kb+16,kb+24)
        // B frag: lane col = lane&15, K in [kb,kb+8)+[kb+16,kb+24) (same as A)
        const int kb  = (lane >> 4) * 8;
        const int kb2 = (lane >> 4) * 16;

        v16bf af[4];
#pragma unroll
        for (int mt = 0; mt < 4; mt++) {
            int rr = wm * 64 + mt * 16 + (lane & 15);
            v8bf a0 = *(const v8bf*)&As[rr][kb];
            v8bf a1 = *(const v8bf*)&As[rr][kb + 16];
            af[mt] = __builtin_shufflevector(a0, a1,
                     0,1,2,3,4,5,6,7,8,9,10,11,12,13,14,15);
        }
#pragma unroll
        for (int nt = 0; nt < 2; nt++) {
            int cc = wn * 32 + nt * 16 + (lane & 15);
            v8bf b0 = *(const v8bf*)&Bs[cc][kb];
            v8bf b1 = *(const v8bf*)&Bs[cc][kb + 16];
            v16bf bv = __builtin_shufflevector(b0, b1,
                       0,1,2,3,4,5,6,7,8,9,10,11,12,13,14,15);
#pragma unroll
            for (int mt = 0; mt < 4; mt++) {
                acc[mt][nt] = __builtin_amdgcn_wmma_f32_16x16x32_bf16(
                    false, af[mt], false, bv, (short)0, acc[mt][nt], false, false);
            }
        }
        __syncthreads();   // LDS consumed; next iteration may overwrite
    }

    // ---- epilogue + store (branch-free per instantiation) ----
#pragma unroll
    for (int mt = 0; mt < 4; mt++) {
#pragma unroll
        for (int nt = 0; nt < 2; nt++) {
            int col = bn + wn * 32 + nt * 16 + (lane & 15);
#pragma unroll
            for (int r = 0; r < 8; r++) {
                int row = bm + wm * 64 + mt * 16 + r + ((lane >> 4) << 3);
                if (row < M && col < N) {
                    float v = acc[mt][nt][r];
                    if (bias) v += bias[col];
                    if (MODE == 1) {
                        v += res[(long)row * ldc + col];
                    } else if (MODE == 2) {
                        v = 0.5f * v * (1.f + erff(v * 0.70710678118f));
                    } else if (MODE == 3) {
                        v = (v * sigmoidf_(v)) * res[(long)row * ldc + col];
                    }
                    Cb[(long)row * ldc + col] = v;
                }
            }
        }
    }
}

// ---------------------------------------------------------------------------
// Elementwise / reduction kernels
// ---------------------------------------------------------------------------

__global__ __launch_bounds__(256)
void rmsnorm_kernel(const float* __restrict__ x, const float* __restrict__ s,
                    float* __restrict__ y, int D)
{
    __shared__ float red[256];
    const int row = blockIdx.x;
    const float* xr = x + (long)row * D;
    float acc = 0.f;
    for (int i = threadIdx.x; i < D; i += 256) { float v = xr[i]; acc += v * v; }
    red[threadIdx.x] = acc; __syncthreads();
    for (int o = 128; o > 0; o >>= 1) {
        if (threadIdx.x < o) red[threadIdx.x] += red[threadIdx.x + o];
        __syncthreads();
    }
    float inv = rsqrtf(red[0] / (float)D + 1e-6f);
    for (int i = threadIdx.x; i < D; i += 256)
        y[(long)row * D + i] = xr[i] * inv * s[i];
}

__global__ __launch_bounds__(256)
void rope_kernel(const float* __restrict__ qkv,
                 float* __restrict__ qr, float* __restrict__ kr)
{
    int idx = blockIdx.x * 256 + threadIdx.x;
    if (idx >= T_TOK * DMODEL) return;
    int t    = idx / DMODEL;
    int dd   = idx % DMODEL;
    int head = dd / HDIM;
    int h    = dd % HDIM;
    int j    = (h < 40) ? h : h - 40;
    float pos = (j < 20) ? (float)(t / 32) : (float)(t % 32);
    int   i   = j % 20;
    float invf = __powf(10000.f, -(float)i / 20.f);
    float ang  = pos * invf;
    float c = __cosf(ang), sn = __sinf(ang);
    const float* qp = qkv + (long)t * (3 * DMODEL) + head * HDIM;
    const float* kp = qp + DMODEL;
    float qo, ko;
    if (h < 40) { qo = qp[h] * c - qp[h + 40] * sn;  ko = kp[h] * c - kp[h + 40] * sn; }
    else        { qo = qp[h - 40] * sn + qp[h] * c;  ko = kp[h - 40] * sn + kp[h] * c; }
    qr[idx] = qo;
    kr[idx] = ko;
}

__global__ __launch_bounds__(256)
void softmax_kernel(float* __restrict__ S, float scale, int use_mask)
{
    __shared__ float red[256];
    const int t    = blockIdx.x;
    const int head = blockIdx.y;
    float* row = S + ((long)head * T_TOK + t) * T_TOK;
    float vals[4];
    float mx = -3.40282347e38f;
#pragma unroll
    for (int i = 0; i < 4; i++) {
        int col = threadIdx.x + i * 256;
        float v = row[col] * scale;
        if (use_mask) {
            int d = t - col; if (d < 0) d = -d;
            if (d > WINDOW_SZ) v = -3.40282347e38f;
        }
        vals[i] = v;
        mx = fmaxf(mx, v);
    }
    red[threadIdx.x] = mx; __syncthreads();
    for (int o = 128; o > 0; o >>= 1) {
        if (threadIdx.x < o) red[threadIdx.x] = fmaxf(red[threadIdx.x], red[threadIdx.x + o]);
        __syncthreads();
    }
    mx = red[0]; __syncthreads();
    float sum = 0.f;
#pragma unroll
    for (int i = 0; i < 4; i++) { vals[i] = __expf(vals[i] - mx); sum += vals[i]; }
    red[threadIdx.x] = sum; __syncthreads();
    for (int o = 128; o > 0; o >>= 1) {
        if (threadIdx.x < o) red[threadIdx.x] += red[threadIdx.x + o];
        __syncthreads();
    }
    float inv = 1.f / red[0];
#pragma unroll
    for (int i = 0; i < 4; i++) row[threadIdx.x + i * 256] = vals[i] * inv;
}

__global__ __launch_bounds__(256)
void permute_kernel(const float* __restrict__ x, float* __restrict__ xp)
{
    const int E = 2 * 14 * 14 * 3;  // 1176
    int idx = blockIdx.x * 256 + threadIdx.x;
    if (idx >= T_TOK * E) return;
    int t = idx / E;
    int e = idx % E;
    int c = e % 3;
    int r = e / 3;
    int p2 = r % 14; r /= 14;
    int p1 = r % 14;
    int tp = r / 14;
    int src = ((c * 2 + tp) * 14 + p1) * 14 + p2;
    xp[idx] = x[(long)t * E + src];
}

// ---------------------------------------------------------------------------
// Host-side orchestration
// ---------------------------------------------------------------------------
struct GemmArgs {
    const float* A; long lda, sA;
    const float* B; long ldb, sB;
    float* C; long ldc, sC;
    const float* bias; const float* res;
    int M, N, K, batch;
};

template <int TRANSB, int MODE>
static void launch_gemm(const GemmArgs& a, hipStream_t stream)
{
    dim3 gr((unsigned)((a.N + 127) / 128), (unsigned)((a.M + 127) / 128),
            (unsigned)a.batch);
    gemm_bf16_wmma<TRANSB, MODE><<<gr, 256, 0, stream>>>(
        a.A, a.lda, a.sA, a.B, a.ldb, a.sB,
        a.C, a.ldc, a.sC, a.bias, a.res, a.M, a.N, a.K);
}

extern "C" void kernel_launch(void* const* d_in, const int* in_sizes, int n_in,
                              void* d_out, int out_size, void* d_ws, size_t ws_size,
                              hipStream_t stream)
{
    (void)in_sizes; (void)n_in; (void)out_size; (void)ws_size;

    const float* x      = (const float*)d_in[0];
    const float* patchw = (const float*)d_in[1];
    const float* n1s    = (const float*)d_in[2];
    const float* n2s    = (const float*)d_in[3];
    const float* wqkv   = (const float*)d_in[4];
    const float* bqkv   = (const float*)d_in[5];
    const float* wproj  = (const float*)d_in[6];
    const float* bproj  = (const float*)d_in[7];
    const float* wg     = (const float*)d_in[8];
    const float* bg     = (const float*)d_in[9];
    const float* wu     = (const float*)d_in[10];
    const float* bu     = (const float*)d_in[11];
    const float* wd     = (const float*)d_in[12];
    const float* bd     = (const float*)d_in[13];
    const float* lnq    = (const float*)d_in[14];
    const float* fc1w   = (const float*)d_in[15];
    const float* fc1b   = (const float*)d_in[16];
    const float* fc2w   = (const float*)d_in[17];
    const float* fc2b   = (const float*)d_in[18];
    float* out = (float*)d_out;

    float* ws  = (float*)d_ws;
    float* xp  = ws;
    float* h   = xp  + (size_t)T_TOK * 1176;
    float* y   = h   + (size_t)T_TOK * DMODEL;
    float* qkv = y   + (size_t)T_TOK * DMODEL;
    float* qr  = qkv + (size_t)T_TOK * 3 * DMODEL;
    float* kr  = qr  + (size_t)T_TOK * DMODEL;
    float* sc  = kr  + (size_t)T_TOK * DMODEL;
    float* ao  = sc  + (size_t)NHEAD * T_TOK * T_TOK;
    float* u   = ao  + (size_t)T_TOK * DMODEL;
    float* g   = u   + (size_t)T_TOK * INTER_F;
    float* f1  = g   + (size_t)T_TOK * INTER_F;

    permute_kernel<<<(T_TOK * 1176 + 255) / 256, 256, 0, stream>>>(x, xp);
    launch_gemm<0, 0>({xp, 1176, 0, patchw, DMODEL, 0, h, DMODEL, 0,
                       nullptr, nullptr, T_TOK, DMODEL, 1176, 1}, stream);

    const float scale = 0.11180339887f;  // 1/sqrt(80)

    for (int d = 0; d < 8; d++) {
        rmsnorm_kernel<<<T_TOK, 256, 0, stream>>>(h, n1s + (long)d * DMODEL, y, DMODEL);
        launch_gemm<0, 0>({y, DMODEL, 0,
                           wqkv + (long)d * DMODEL * 3 * DMODEL, 3 * DMODEL, 0,
                           qkv, 3 * DMODEL, 0,
                           bqkv + (long)d * 3 * DMODEL, nullptr,
                           T_TOK, 3 * DMODEL, DMODEL, 1}, stream);
        rope_kernel<<<(T_TOK * DMODEL + 255) / 256, 256, 0, stream>>>(qkv, qr, kr);

        // S = Q K^T, batched over heads (B transposed)
        launch_gemm<1, 0>({qr, DMODEL, HDIM, kr, DMODEL, HDIM,
                           sc, T_TOK, (long)T_TOK * T_TOK,
                           nullptr, nullptr, T_TOK, T_TOK, HDIM, NHEAD}, stream);

        int use_mask = (d != 3 && d != 7);
        softmax_kernel<<<dim3(T_TOK, NHEAD), 256, 0, stream>>>(sc, scale, use_mask);

        // O = P V, batched over heads
        launch_gemm<0, 0>({sc, T_TOK, (long)T_TOK * T_TOK,
                           qkv + 2 * DMODEL, 3 * DMODEL, HDIM,
                           ao, DMODEL, HDIM,
                           nullptr, nullptr, T_TOK, HDIM, T_TOK, NHEAD}, stream);

        // h = h + O @ wproj + bproj
        launch_gemm<0, 1>({ao, DMODEL, 0,
                           wproj + (long)d * DMODEL * DMODEL, DMODEL, 0,
                           h, DMODEL, 0,
                           bproj + (long)d * DMODEL, h,
                           T_TOK, DMODEL, DMODEL, 1}, stream);

        rmsnorm_kernel<<<T_TOK, 256, 0, stream>>>(h, n2s + (long)d * DMODEL, y, DMODEL);
        launch_gemm<0, 0>({y, DMODEL, 0,
                           wu + (long)d * DMODEL * INTER_F, INTER_F, 0,
                           u, INTER_F, 0,
                           bu + (long)d * INTER_F, nullptr,
                           T_TOK, INTER_F, DMODEL, 1}, stream);
        launch_gemm<0, 3>({y, DMODEL, 0,
                           wg + (long)d * DMODEL * INTER_F, INTER_F, 0,
                           g, INTER_F, 0,
                           bg + (long)d * INTER_F, u,
                           T_TOK, INTER_F, DMODEL, 1}, stream);
        launch_gemm<0, 1>({g, INTER_F, 0,
                           wd + (long)d * INTER_F * DMODEL, DMODEL, 0,
                           h, DMODEL, 0,
                           bd + (long)d * DMODEL, h,
                           T_TOK, DMODEL, INTER_F, 1}, stream);
    }

    rmsnorm_kernel<<<T_TOK, 256, 0, stream>>>(h, lnq, y, DMODEL);
    launch_gemm<0, 2>({y, 5120, 0, fc1w, 5120, 0, f1, 5120, 0,
                       fc1b, nullptr, 256, 5120, 5120, 1}, stream);
    launch_gemm<0, 0>({f1, 5120, 0, fc2w, 2048, 0, out, 2048, 0,
                       fc2b, nullptr, 256, 2048, 5120, 1}, stream);
}
